// MoE_56934086476111
// MI455X (gfx1250) — matrix operand order repaired and checked by reading, not compiled, not run on hardware
//
#include <hip/hip_runtime.h>
#include <hip/hip_bf16.h>

// ---------------- problem constants ----------------
constexpr int DIM    = 1024;
constexpr int HIDDEN = 2816;
constexpr int NE     = 8;      // experts
constexpr int TT     = 4096;   // tokens (2*2048)
constexpr int TILE_M = 32;     // tokens per FFN block (2 x 16-row WMMA tiles)

// ---------------- vector types ----------------
typedef __attribute__((ext_vector_type(16))) __bf16 v16bf;
typedef __attribute__((ext_vector_type(8)))  __bf16 v8bf;
typedef __attribute__((ext_vector_type(8)))  float  v8f;

// ---------------- workspace layout (bytes) ----------------
constexpr size_t SZ_XB  = (size_t)TT * DIM * 2;           // x in bf16
constexpr size_t SZ_W   = (size_t)NE * HIDDEN * DIM * 2;  // one weight set bf16
constexpr size_t OFF_XB  = 0;
constexpr size_t OFF_W1B = OFF_XB + SZ_XB;
constexpr size_t OFF_W3B = OFF_W1B + SZ_W;
constexpr size_t OFF_W2T = OFF_W3B + SZ_W;                // W2 transposed [e][n][h] bf16
constexpr size_t OFF_CMB = OFF_W2T + SZ_W;                // combine (TT x NE) f32
constexpr size_t OFF_TIDX= OFF_CMB + (size_t)TT * NE * 4; // token index lists
constexpr size_t OFF_TW  = OFF_TIDX + (size_t)NE * TT * 4;// token weight lists
constexpr size_t OFF_CNT = OFF_TW  + (size_t)NE * TT * 4; // counts (NE ints)

// ---------------- fp32 -> bf16 convert ----------------
__global__ void f32_to_bf16_kernel(const float* __restrict__ s,
                                   __bf16* __restrict__ d, int n) {
    int i = blockIdx.x * blockDim.x + threadIdx.x;
    if (i < n) d[i] = (__bf16)s[i];
}

// ---------------- W2 transpose to [e][n][h] bf16 ----------------
// grid: (HIDDEN/32, DIM/32, NE), block: (32, 8)
__global__ __launch_bounds__(256) void w2_transpose_kernel(const float* __restrict__ src,
                                                           __bf16* __restrict__ dst) {
    __shared__ float tile[32][33];
    int e  = blockIdx.z;
    int h0 = blockIdx.x * 32;
    int n0 = blockIdx.y * 32;
    #pragma unroll
    for (int i = 0; i < 4; ++i) {
        int h = h0 + threadIdx.y + i * 8;
        tile[threadIdx.y + i * 8][threadIdx.x] =
            src[((size_t)e * HIDDEN + h) * DIM + n0 + threadIdx.x];
    }
    __syncthreads();
    #pragma unroll
    for (int i = 0; i < 4; ++i) {
        int n = n0 + threadIdx.y + i * 8;
        dst[((size_t)e * DIM + n) * HIDDEN + h0 + threadIdx.x] =
            (__bf16)tile[threadIdx.x][threadIdx.y + i * 8];
    }
}

// ---------------- gating: softmax + top-2 + renorm ----------------
// one wave (32 lanes) per token; grid = TT/8, block = 256 (8 waves)
__global__ __launch_bounds__(256) void gate_kernel(const float* __restrict__ x,
                                                   const float* __restrict__ gw,
                                                   float* __restrict__ combine) {
    int wave = threadIdx.x >> 5, lane = threadIdx.x & 31;
    int t = blockIdx.x * 8 + wave;
    const float* xr = x + (size_t)t * DIM;
    float acc[NE];
    #pragma unroll
    for (int e = 0; e < NE; ++e) acc[e] = 0.f;
    for (int k = lane; k < DIM; k += 32) {
        float xv = xr[k];
        #pragma unroll
        for (int e = 0; e < NE; ++e) acc[e] += xv * gw[e * DIM + k];
    }
    #pragma unroll
    for (int off = 16; off > 0; off >>= 1) {
        #pragma unroll
        for (int e = 0; e < NE; ++e) acc[e] += __shfl_down(acc[e], off);
    }
    if (lane == 0) {
        float mx = acc[0];
        #pragma unroll
        for (int e = 1; e < NE; ++e) mx = fmaxf(mx, acc[e]);
        float p[NE], s = 0.f;
        #pragma unroll
        for (int e = 0; e < NE; ++e) { p[e] = __expf(acc[e] - mx); s += p[e]; }
        float inv = 1.f / s;
        #pragma unroll
        for (int e = 0; e < NE; ++e) p[e] *= inv;
        int e1 = 0;
        #pragma unroll
        for (int e = 1; e < NE; ++e) if (p[e] > p[e1]) e1 = e;
        int e2 = (e1 == 0) ? 1 : 0;
        #pragma unroll
        for (int e = 0; e < NE; ++e) if (e != e1 && p[e] > p[e2]) e2 = e;
        float sw = 1.f / (p[e1] + p[e2]);
        #pragma unroll
        for (int e = 0; e < NE; ++e)
            combine[(size_t)t * NE + e] =
                (e == e1) ? p[e1] * sw : ((e == e2) ? p[e2] * sw : 0.f);
    }
}

// ---------------- deterministic wave-parallel compaction ----------------
// one wave per expert; ballot/popcount keeps token order (bit-identical lists)
__global__ void compact_kernel(const float* __restrict__ combine,
                               int* __restrict__ tokidx, float* __restrict__ tokw,
                               int* __restrict__ counts) {
    int e = blockIdx.x;
    int lane = threadIdx.x;
    int c = 0;
    for (int t0 = 0; t0 < TT; t0 += 32) {
        float w = combine[(size_t)(t0 + lane) * NE + e];
        unsigned int mask = __builtin_amdgcn_ballot_w32(w > 0.f);
        int pos = __popc(mask & ((1u << lane) - 1u));
        if (w > 0.f) {
            tokidx[e * TT + c + pos] = t0 + lane;
            tokw  [e * TT + c + pos] = w;
        }
        c += __popc(mask);
    }
    if (lane == 0) counts[e] = c;
}

__global__ void zero_kernel(float* __restrict__ o, int n) {
    int i = blockIdx.x * blockDim.x + threadIdx.x;
    if (i < n) o[i] = 0.f;
}

// ---------------- fused SwiGLU FFN per expert-tile ----------------
// grid: (TT/TILE_M, NE), block: 256 threads = 8 waves.
// Each block handles 32 routed tokens of one expert (two 16-row WMMA tiles
// sharing every B fragment), fully fused:
//   h = silu(x W1^T) * (x W3^T); out += (h W2) * combine_weight
__global__ __launch_bounds__(256) void ffn_kernel(
    const __bf16* __restrict__ xb,
    const __bf16* __restrict__ w1b,
    const __bf16* __restrict__ w3b,
    const __bf16* __restrict__ w2t,   // [e][n][h]
    const int*    __restrict__ tokidx,
    const float*  __restrict__ tokw,
    const int*    __restrict__ counts,
    float*        __restrict__ out) {

    int e    = blockIdx.y;
    int cnt  = counts[e];
    int tile = blockIdx.x;
    if (tile * TILE_M >= cnt) return;

    __shared__ __align__(32) __bf16 hb[TILE_M][128];  // 8 KB: hidden chunk, bf16
    __shared__ int   rows[TILE_M];
    __shared__ float cws[TILE_M];

    int tid  = threadIdx.x;
    int wave = tid >> 5;
    int lane = tid & 31;

    if (tid < TILE_M) {
        int i = tile * TILE_M + tid;
        if (i < cnt) { rows[tid] = tokidx[e * TT + i]; cws[tid] = tokw[e * TT + i]; }
        else         { rows[tid] = 0;                  cws[tid] = 0.f; }
    }
    __syncthreads();

    // WMMA lane mapping (contiguous k order, same for A and B)
    const int m    = lane & 15;         // A-matrix row within a 16-tile
    const int aoff = (lane >> 4) * 16;  // A-frag: 16 contiguous K per lane
    const int bn   = lane & 15;         // B-frag column
    const int bk   = (lane >> 4) * 16;  // B-frag: 16 contiguous K per lane

    // per-lane A-row base pointers (tokens; identical across waves -> WGP$ hits)
    const __bf16* pa0 = xb + (size_t)rows[m]      * DIM;
    const __bf16* pa1 = xb + (size_t)rows[16 + m] * DIM;

    v8f acc[16];                        // [0..7]: tokens 0-15, [8..15]: tokens 16-31
    #pragma unroll
    for (int j = 0; j < 16; ++j) acc[j] = {};

    for (int hc = 0; hc < HIDDEN / 128; ++hc) {
        int h0  = hc * 128;
        int myh = h0 + wave * 16;
        const __bf16* b1p = w1b + ((size_t)e * HIDDEN + myh + bn) * DIM;
        const __bf16* b3p = w3b + ((size_t)e * HIDDEN + myh + bn) * DIM;

        // ---- stage A: g1 = x W1_rows^T ; g3 = x W3_rows^T (2 token tiles x 16 hidden)
        v8f g1a = {}, g3a = {}, g1b = {}, g3b = {};
        for (int k0 = 0; k0 < DIM; k0 += 32) {
            v16bf af0 = __builtin_shufflevector(
                *(const v8bf*)&pa0[k0 + aoff], *(const v8bf*)&pa0[k0 + aoff + 8],
                0,1,2,3,4,5,6,7,8,9,10,11,12,13,14,15);
            v16bf af1 = __builtin_shufflevector(
                *(const v8bf*)&pa1[k0 + aoff], *(const v8bf*)&pa1[k0 + aoff + 8],
                0,1,2,3,4,5,6,7,8,9,10,11,12,13,14,15);
            v16bf b1f = *(const v16bf*)&b1p[k0 + bk];
            v16bf b3f = *(const v16bf*)&b3p[k0 + bk];
            g1a = __builtin_amdgcn_wmma_f32_16x16x32_bf16(false, af0, false, b1f,
                                                          (short)0, g1a, false, false);
            g1b = __builtin_amdgcn_wmma_f32_16x16x32_bf16(false, af1, false, b1f,
                                                          (short)0, g1b, false, false);
            g3a = __builtin_amdgcn_wmma_f32_16x16x32_bf16(false, af0, false, b3f,
                                                          (short)0, g3a, false, false);
            g3b = __builtin_amdgcn_wmma_f32_16x16x32_bf16(false, af1, false, b3f,
                                                          (short)0, g3b, false, false);
        }

        // ---- SiLU gate -> hb (bf16), 32 tokens x 128 hidden per chunk
        #pragma unroll
        for (int j = 0; j < 8; ++j) {
            int mrow = j + (lane >> 4) * 8;
            float v1 = g1a[j];
            hb[mrow][wave * 16 + bn]      = (__bf16)((v1 / (1.f + __expf(-v1))) * g3a[j]);
            float v2 = g1b[j];
            hb[16 + mrow][wave * 16 + bn] = (__bf16)((v2 / (1.f + __expf(-v2))) * g3b[j]);
        }
        __syncthreads();

        // ---- stage B: acc += h_chunk @ W2_chunk  (wave owns 128 out columns)
        #pragma unroll
        for (int k0 = 0; k0 < 128; k0 += 32) {
            v16bf af0 = __builtin_shufflevector(
                *(const v8bf*)&hb[m][k0 + aoff], *(const v8bf*)&hb[m][k0 + aoff + 8],
                0,1,2,3,4,5,6,7,8,9,10,11,12,13,14,15);
            v16bf af1 = __builtin_shufflevector(
                *(const v8bf*)&hb[16 + m][k0 + aoff], *(const v8bf*)&hb[16 + m][k0 + aoff + 8],
                0,1,2,3,4,5,6,7,8,9,10,11,12,13,14,15);
            #pragma unroll
            for (int jt = 0; jt < 8; ++jt) {
                int col = wave * 128 + jt * 16 + bn;
                const __bf16* bp = w2t + ((size_t)e * DIM + col) * HIDDEN + h0 + k0 + bk;
                v16bf bf_ = *(const v16bf*)bp;
                acc[jt]     = __builtin_amdgcn_wmma_f32_16x16x32_bf16(false, af0, false, bf_,
                                                                      (short)0, acc[jt],
                                                                      false, false);
                acc[8 + jt] = __builtin_amdgcn_wmma_f32_16x16x32_bf16(false, af1, false, bf_,
                                                                      (short)0, acc[8 + jt],
                                                                      false, false);
            }
        }
        __syncthreads();
    }

    // ---- epilogue: scale by gate weight, accumulate into output
    #pragma unroll
    for (int jt = 0; jt < 8; ++jt) {
        #pragma unroll
        for (int j = 0; j < 8; ++j) {
            int mrow = j + (lane >> 4) * 8;
            int col  = wave * 128 + jt * 16 + bn;
            atomicAdd(&out[(size_t)rows[mrow] * DIM + col],      acc[jt][j]     * cws[mrow]);
            atomicAdd(&out[(size_t)rows[16 + mrow] * DIM + col], acc[8 + jt][j] * cws[16 + mrow]);
        }
    }
}

// ---------------- launcher ----------------
extern "C" void kernel_launch(void* const* d_in, const int* in_sizes, int n_in,
                              void* d_out, int out_size, void* d_ws, size_t ws_size,
                              hipStream_t stream) {
    const float* x  = (const float*)d_in[0];
    const float* w1 = (const float*)d_in[1];
    const float* w2 = (const float*)d_in[2];
    const float* w3 = (const float*)d_in[3];
    const float* gw = (const float*)d_in[4];
    float* out = (float*)d_out;

    char* ws = (char*)d_ws;
    __bf16* xb   = (__bf16*)(ws + OFF_XB);
    __bf16* w1b  = (__bf16*)(ws + OFF_W1B);
    __bf16* w3b  = (__bf16*)(ws + OFF_W3B);
    __bf16* w2t  = (__bf16*)(ws + OFF_W2T);
    float*  cmb  = (float*) (ws + OFF_CMB);
    int*    tidx = (int*)   (ws + OFF_TIDX);
    float*  tw   = (float*) (ws + OFF_TW);
    int*    cnts = (int*)   (ws + OFF_CNT);

    const int NX = TT * DIM;
    const int NW = NE * HIDDEN * DIM;

    f32_to_bf16_kernel<<<(NX + 255) / 256, 256, 0, stream>>>(x,  xb,  NX);
    f32_to_bf16_kernel<<<(NW + 255) / 256, 256, 0, stream>>>(w1, w1b, NW);
    f32_to_bf16_kernel<<<(NW + 255) / 256, 256, 0, stream>>>(w3, w3b, NW);
    w2_transpose_kernel<<<dim3(HIDDEN / 32, DIM / 32, NE), dim3(32, 8), 0, stream>>>(w2, w2t);

    gate_kernel<<<TT / 8, 256, 0, stream>>>(x, gw, cmb);
    compact_kernel<<<NE, 32, 0, stream>>>(cmb, tidx, tw, cnts);
    zero_kernel<<<(NX + 255) / 256, 256, 0, stream>>>(out, NX);

    ffn_kernel<<<dim3(TT / TILE_M, NE), 256, 0, stream>>>(xb, w1b, w3b, w2t,
                                                          tidx, tw, cnts, out);
}
